// NonLocalBlock3DWithDownsampling_91001767067855
// MI455X (gfx1250) — compile-verified, matrix operand order checked by reading, not run on hardware
//
#include <hip/hip_runtime.h>
#include <hip/hip_bf16.h>
#include <math.h>

typedef _Float16 f16_t;
typedef __attribute__((ext_vector_type(16))) _Float16 v16h;
typedef __attribute__((ext_vector_type(8)))  _Float16 v8h;
typedef __attribute__((ext_vector_type(8)))  float    v8f;

// ---------------------------------------------------------------------------
// Problem constants
// ---------------------------------------------------------------------------
#define NB   2          // batch
#define CIN  256        // input channels
#define ICH  128        // inner channels
#define THW  16384      // T*H*W   (queries)
#define TKW  4096       // T*Hd*Wd (keys)
#define GPAD 160        // 131 padded to 5*32 for WMMA K-steps
#define OCH  256        // output channels

#define AS_LOCAL __attribute__((address_space(3)))

// ---------------------------------------------------------------------------
// Async global->LDS copy (CDNA5 GLOBAL_LOAD_ASYNC_TO_LDS_B128, ASYNCcnt).
// Inline asm: the clang builtin's pointer parameters use the cuda_device
// LangAS which cannot be spelled from HIP source, so we emit the instruction
// directly.  vdst = LDS byte offset VGPR, vaddr = 64-bit global address pair.
// ---------------------------------------------------------------------------
__device__ __forceinline__ void async_copy16(const f16_t* g, f16_t* l) {
#if defined(__gfx1250__)
  unsigned lds_off = (unsigned)(size_t)(AS_LOCAL const void*)l;
  asm volatile("global_load_async_to_lds_b128 %0, %1, off"
               :: "v"(lds_off), "v"(g) : "memory");
#else
  *(v8h*)l = *(const v8h*)g;
#endif
}

__device__ __forceinline__ void wait_async() {
#if defined(__gfx1250__)
  asm volatile("s_wait_asynccnt 0x0" ::: "memory");
#endif
}

// ---------------------------------------------------------------------------
// WMMA helpers (CDNA5 / gfx1250, wave32)
// ---------------------------------------------------------------------------
__device__ __forceinline__ v8f wmma_f16(v16h a, v16h b, v8f c) {
  // v_wmma_f32_16x16x32_f16  D = A*B + C
  return __builtin_amdgcn_wmma_f32_16x16x32_f16(
      /*neg_a=*/false, a, /*neg_b=*/false, b,
      /*c_mod=*/(short)0, c, /*reuse_a=*/false, /*reuse_b=*/false);
}

// Load a 16x32 f16 A/B fragment. `p` points at (row_or_col base)*ldk + k0 in a
// row/col-major f16 buffer with K contiguous.  Per ISA layout:
//   lanes 0-15 : K = k0+0..7   and k0+16..23
//   lanes16-31 : K = k0+8..15  and k0+24..31
__device__ __forceinline__ v16h frag_ld(const f16_t* p, int lane) {
  int kb = (lane >> 4) * 8;
  v16h f;
  *(v8h*)&f          = *(const v8h*)(p + kb);
  *(((v8h*)&f) + 1)  = *(const v8h*)(p + kb + 16);
  return f;
}

__device__ __forceinline__ float red_max16(float v) {
#pragma unroll
  for (int m = 1; m < 16; m <<= 1) v = fmaxf(v, __shfl_xor(v, m));
  return v;
}
__device__ __forceinline__ float red_sum16(float v) {
#pragma unroll
  for (int m = 1; m < 16; m <<= 1) v += __shfl_xor(v, m);
  return v;
}

// ---------------------------------------------------------------------------
// K1: 3x3/stride2/pad1 average pool (count_include_pad semantics folded into
//     per-position bias scale later), f32 -> f16.  xp layout: [N*C][4096]
// ---------------------------------------------------------------------------
__global__ __launch_bounds__(256) void pool_kernel(const float* __restrict__ x,
                                                   f16_t* __restrict__ xp) {
  int idx = blockIdx.x * 256 + threadIdx.x;
  if (idx >= NB * CIN * TKW) return;
  int key = idx & (TKW - 1);
  int nc  = idx >> 12;
  int t = key >> 10, sp = key & 1023, hd = sp >> 5, wd = sp & 31;
  const float* src = x + ((size_t)nc * 4 + t) * 4096;  // [64][64] plane
  int h0 = 2 * hd - 1, w0 = 2 * wd - 1;
  float s = 0.f;
#pragma unroll
  for (int dh = 0; dh < 3; ++dh) {
    int h = h0 + dh;
    if ((unsigned)h < 64u) {
#pragma unroll
      for (int dw = 0; dw < 3; ++dw) {
        int w = w0 + dw;
        if ((unsigned)w < 64u) s += src[h * 64 + w];
      }
    }
  }
  xp[idx] = (f16_t)(s * (1.f / 9.f));
}

// ---------------------------------------------------------------------------
// K2: fill grid channels (128..130) + zero pad (131..159) of Gt, and convert
//     W_w [256,131] f32 -> W2h [256,160] f16 (zero padded).
// ---------------------------------------------------------------------------
__global__ __launch_bounds__(256) void prep_extra_kernel(const float* __restrict__ W_w,
                                                         f16_t* __restrict__ Gt,
                                                         f16_t* __restrict__ W2h) {
  int idx = blockIdx.x * 256 + threadIdx.x;
  const int GTE = NB * 32 * TKW;
  if (idx < GTE) {
    int key = idx & (TKW - 1);
    int r   = (idx >> 12) & 31;
    int n   = idx >> 17;
    int t = key >> 10, hd = (key >> 5) & 31, wd = key & 31;
    float v = 0.f;
    if (r == 0)      v = ((float)t / 1.5f - 1.f) * 0.1f;   // tg * t_scale
    else if (r == 1) v = (float)hd / 15.5f - 1.f;          // yg
    else if (r == 2) v = (float)wd / 7.75f - 2.f;          // xg
    Gt[((size_t)n * GPAD + 128 + r) * TKW + key] = (f16_t)v;
  } else {
    int j = idx - GTE;
    if (j < OCH * GPAD) {
      int oc = j / GPAD, k = j % GPAD;
      W2h[j] = (f16_t)(k < 131 ? W_w[oc * 131 + k] : 0.f);
    }
  }
}

// ---------------------------------------------------------------------------
// K3/K4/K5: 1x1 conv as WMMA GEMM  D[ic, pos] = sum_c W[ic,c] * X[c,pos] + b
//   VAR 0 : Q  : in = x  (f32, S=16384), out Q[(n*THW+pos)*128 + ic]
//   VAR 1 : Phi: in = xp (f16, S=4096),  out Phi[(n*TKW+pos)*128 + ic], bias*cnt/9
//   VAR 2 : G  : in = xp (f16, S=4096),  out Gt[(n*160+ic)*TKW + pos], bias*cnt/9
// Workgroup: 128 positions x 128 ic; 8 waves = one 16-ic strip each.
// (Staging converts dtype / transposes, so it cannot use raw async copies.)
// ---------------------------------------------------------------------------
template <int VAR>
__global__ __launch_bounds__(256) void conv_kernel(const void* __restrict__ xin,
                                                   const float* __restrict__ w,
                                                   const float* __restrict__ b,
                                                   f16_t* __restrict__ outp) {
  const int S = (VAR == 0) ? THW : TKW;
  __shared__ __align__(16) f16_t lw[ICH * 32];   // weights [ic][32]
  __shared__ __align__(16) f16_t lx[128 * 32];   // activ   [pos][32] (transposed)
  int tid = threadIdx.x, lane = tid & 31, wv = tid >> 5;
  int n = blockIdx.y;
  int pbase = blockIdx.x * 128;
  const float* xf = (VAR == 0) ? ((const float*)xin + (size_t)n * CIN * THW) : nullptr;
  const f16_t* xh = (VAR != 0) ? ((const f16_t*)xin + (size_t)n * CIN * TKW) : nullptr;

  v8f acc[8] = {};
  for (int cb = 0; cb < CIN; cb += 32) {
    __syncthreads();
    {  // stage weight chunk: thread -> 16 contiguous c of one ic row
      int ic = tid >> 1, hf = (tid & 1) * 16;
      const float* wp = w + ic * CIN + cb + hf;
      f16_t* dp = lw + ic * 32 + hf;
#pragma unroll
      for (int i = 0; i < 16; ++i) dp[i] = (f16_t)wp[i];
    }
    // stage activation chunk transposed: lx[pos][c]
    for (int u = tid; u < 4096; u += 256) {
      int c = u >> 7, p = u & 127;
      float val = (VAR == 0) ? xf[(size_t)(cb + c) * S + pbase + p]
                             : (float)xh[(size_t)(cb + c) * S + pbase + p];
      lx[p * 32 + c] = (f16_t)val;
    }
    __syncthreads();
    v16h a = frag_ld(lw + (wv * 16 + (lane & 15)) * 32, lane);
#pragma unroll
    for (int s = 0; s < 8; ++s) {
      v16h bb = frag_ld(lx + (s * 16 + (lane & 15)) * 32, lane);
      acc[s] = wmma_f16(a, bb, acc[s]);
    }
  }
  // epilogue: bias (+ avgpool zero-pad valid-count scaling), store f16
  int col = lane & 15, hi = lane >> 4;
#pragma unroll
  for (int s = 0; s < 8; ++s) {
    int p = pbase + s * 16 + col;
    float bscale = 1.f;
    if (VAR != 0) {
      int hd = (p >> 5) & 31, wd = p & 31;
      bscale = ((hd == 0) ? 2.f : 3.f) * ((wd == 0) ? 2.f : 3.f) * (1.f / 9.f);
    }
#pragma unroll
    for (int v = 0; v < 8; ++v) {
      int ic = wv * 16 + v + 8 * hi;
      float val = acc[s][v] + b[ic] * bscale;
      if (VAR == 2)      outp[((size_t)n * GPAD + ic) * TKW + p] = (f16_t)val;
      else if (VAR == 1) outp[((size_t)n * TKW + p) * ICH + ic]  = (f16_t)val;
      else               outp[((size_t)n * THW + p) * ICH + ic]  = (f16_t)val;
    }
  }
}

// ---------------------------------------------------------------------------
// K6: flash attention.  Per wave: 16 queries; loop 4096 keys in chunks of 64.
//     S = Q*Phi^T (WMMA), online softmax (16-lane xor reductions match C/D
//     fragment layout), P through per-wave LDS (layout conversion), O += P*G.
//     Operand staging uses GLOBAL_LOAD_ASYNC_TO_LDS_B128 (ASYNCcnt).
//     Output Ybuf [N,16384,160] f16 (normalized).
// ---------------------------------------------------------------------------
__global__ __launch_bounds__(256) void attn_kernel(const f16_t* __restrict__ Qbuf,
                                                   const f16_t* __restrict__ Phi,
                                                   const f16_t* __restrict__ Gt,
                                                   f16_t* __restrict__ Ybuf) {
  __shared__ __align__(16) f16_t sphi[64 * ICH];     // [key][c]   16 KB
  __shared__ __align__(16) f16_t sgt[GPAD * 64];     // [oc][key]  20 KB
  __shared__ __align__(16) f16_t sP[8][16 * 64];     // per-wave P 16 KB
  int tid = threadIdx.x, lane = tid & 31, wv = tid >> 5;
  int n = blockIdx.y;
  int qbase = blockIdx.x * 128;
  int qrow = qbase + wv * 16 + (lane & 15);
  const f16_t* qp = Qbuf + ((size_t)n * THW + qrow) * ICH;
  v16h qf[4];
#pragma unroll
  for (int k = 0; k < 4; ++k) qf[k] = frag_ld(qp + 32 * k, lane);

  const f16_t* phin = Phi + (size_t)n * TKW * ICH;
  const f16_t* gtn  = Gt + (size_t)n * GPAD * TKW;
  f16_t* pw = &sP[wv][0];

  v8f o[10] = {};
  float l[8], mx[8];
#pragma unroll
  for (int v = 0; v < 8; ++v) { l[v] = 0.f; mx[v] = -3.0e38f; }

  for (int kc = 0; kc < TKW; kc += 64) {
    __syncthreads();
    // stage phi chunk [64][128] (contiguous 16 KB) via async global->LDS
    for (int u = tid; u < 1024; u += 256)
      async_copy16(phin + (size_t)kc * ICH + u * 8, sphi + u * 8);
    // stage G chunk [160][64] via async global->LDS
    for (int u = tid; u < 1280; u += 256) {
      int oc = u >> 3, off = (u & 7) * 8;
      async_copy16(gtn + (size_t)oc * TKW + kc + off, sgt + oc * 64 + off);
    }
    wait_async();
    __syncthreads();
    if (kc + 64 < TKW) {  // hint next chunk toward caches (global_prefetch_b8)
      __builtin_prefetch(phin + (size_t)(kc + 64) * ICH + lane * 64, 0, 0);
      __builtin_prefetch(gtn + (size_t)lane * TKW + kc + 64, 0, 0);
    }
    // ---- S = Q * Phi^T : 4 key sub-tiles x 4 K-steps = 16 WMMAs
    v8f sfr[4];
#pragma unroll
    for (int s = 0; s < 4; ++s) {
      v8f c = {};
#pragma unroll
      for (int kk = 0; kk < 4; ++kk) {
        v16h bb = frag_ld(sphi + (s * 16 + (lane & 15)) * ICH + kk * 32, lane);
        c = wmma_f16(qf[kk], bb, c);
      }
      sfr[s] = c;
    }
    // ---- online softmax
    float nmax[8], corr[8], rs[8];
#pragma unroll
    for (int v = 0; v < 8; ++v) {
      float m = fmaxf(fmaxf(sfr[0][v], sfr[1][v]), fmaxf(sfr[2][v], sfr[3][v]));
      m = red_max16(m);
      nmax[v] = fmaxf(mx[v], m);
      corr[v] = __expf(mx[v] - nmax[v]);
      mx[v] = nmax[v];
      rs[v] = 0.f;
    }
#pragma unroll
    for (int s = 0; s < 4; ++s) {
#pragma unroll
      for (int v = 0; v < 8; ++v) {
        float p0 = __expf(sfr[s][v] - nmax[v]);
        rs[v] += p0;
        // D layout (row = v+8*hi, col = lane&15) -> row-major P for A frags
        pw[(v + 8 * (lane >> 4)) * 64 + s * 16 + (lane & 15)] = (f16_t)p0;
      }
    }
#pragma unroll
    for (int v = 0; v < 8; ++v) l[v] = l[v] * corr[v] + red_sum16(rs[v]);
#pragma unroll
    for (int t = 0; t < 10; ++t)
#pragma unroll
      for (int v = 0; v < 8; ++v) o[t][v] *= corr[v];
    // ---- O += P * G : 10 oc tiles x 2 K-steps = 20 WMMAs
#pragma unroll
    for (int t = 0; t < 10; ++t) {
#pragma unroll
      for (int ks = 0; ks < 2; ++ks) {
        v16h a  = frag_ld(pw + (lane & 15) * 64 + ks * 32, lane);
        v16h bb = frag_ld(sgt + (t * 16 + (lane & 15)) * 64 + ks * 32, lane);
        o[t] = wmma_f16(a, bb, o[t]);
      }
    }
  }
  // ---- normalize + store Ybuf [q][160]
  int col = lane & 15, hi = lane >> 4;
  f16_t* yo = Ybuf + ((size_t)n * THW + qbase + wv * 16) * GPAD;
#pragma unroll
  for (int t = 0; t < 10; ++t)
#pragma unroll
    for (int v = 0; v < 8; ++v)
      yo[(size_t)(v + 8 * hi) * GPAD + t * 16 + col] = (f16_t)(o[t][v] / l[v]);
}

// ---------------------------------------------------------------------------
// K7: final projection out[n,oc,q] = sum_k Y[q,k]*W2[oc,k] + W_b[oc]  (f32 out)
//     Workgroup: 64 queries x 128 ocs; both operands async-staged in LDS.
// ---------------------------------------------------------------------------
__global__ __launch_bounds__(256) void proj_kernel(const f16_t* __restrict__ Ybuf,
                                                   const f16_t* __restrict__ W2h,
                                                   const float* __restrict__ W_b,
                                                   float* __restrict__ outp) {
  __shared__ __align__(16) f16_t sy[64 * GPAD];    // 20 KB
  __shared__ __align__(16) f16_t sw[128 * GPAD];   // 40 KB
  int tid = threadIdx.x, lane = tid & 31, wv = tid >> 5;
  int n = blockIdx.z;
  int qbase = blockIdx.x * 64;
  int ocbase = blockIdx.y * 128;
  const f16_t* yp = Ybuf + ((size_t)n * THW + qbase) * GPAD;
  for (int u = tid; u < 1280; u += 256) async_copy16(yp + u * 8, sy + u * 8);
  const f16_t* wp = W2h + (size_t)ocbase * GPAD;
  for (int u = tid; u < 2560; u += 256) async_copy16(wp + u * 8, sw + u * 8);
  wait_async();
  __syncthreads();

  int col = lane & 15, hi = lane >> 4;
  float bias = W_b[ocbase + wv * 16 + col];
#pragma unroll
  for (int qs = 0; qs < 4; ++qs) {
    v8f c = {};
#pragma unroll
    for (int kf = 0; kf < 5; ++kf) {
      v16h a  = frag_ld(sy + (qs * 16 + (lane & 15)) * GPAD + kf * 32, lane);
      v16h bb = frag_ld(sw + (wv * 16 + (lane & 15)) * GPAD + kf * 32, lane);
      c = wmma_f16(a, bb, c);
    }
    float* op = outp + ((size_t)n * OCH + ocbase + wv * 16 + col) * THW + qbase + qs * 16;
#pragma unroll
    for (int v = 0; v < 8; ++v) op[v + 8 * hi] = c[v] + bias;
  }
}

// ---------------------------------------------------------------------------
// Host launch
// ---------------------------------------------------------------------------
extern "C" void kernel_launch(void* const* d_in, const int* in_sizes, int n_in,
                              void* d_out, int out_size, void* d_ws, size_t ws_size,
                              hipStream_t stream) {
  const float* x       = (const float*)d_in[0];
  const float* theta_w = (const float*)d_in[1];
  const float* theta_b = (const float*)d_in[2];
  const float* phi_w   = (const float*)d_in[3];
  const float* phi_b   = (const float*)d_in[4];
  const float* g_w     = (const float*)d_in[5];
  const float* g_b     = (const float*)d_in[6];
  const float* W_w     = (const float*)d_in[7];
  const float* W_b     = (const float*)d_in[8];
  float* out = (float*)d_out;

  char* ws = (char*)d_ws;
  f16_t* Qbuf = (f16_t*)(ws + 0);          //  8.0 MB  [N,16384,128]
  f16_t* Phi  = (f16_t*)(ws + 8388608);    //  2.0 MB  [N,4096,128]
  f16_t* Gt   = (f16_t*)(ws + 10485760);   //  2.5 MB  [N,160,4096]
  f16_t* xp   = (f16_t*)(ws + 13107200);   //  4.0 MB  [N*C,4096]
  f16_t* Ybuf = (f16_t*)(ws + 17301504);   // 10.0 MB  [N,16384,160]
  f16_t* W2h  = (f16_t*)(ws + 27787264);   // 80 KB    [256,160]

  pool_kernel<<<(NB * CIN * TKW) / 256, 256, 0, stream>>>(x, xp);
  prep_extra_kernel<<<(NB * 32 * TKW + OCH * GPAD + 255) / 256, 256, 0, stream>>>(W_w, Gt, W2h);
  conv_kernel<0><<<dim3(THW / 128, NB), 256, 0, stream>>>(x,  theta_w, theta_b, Qbuf);
  conv_kernel<1><<<dim3(TKW / 128, NB), 256, 0, stream>>>(xp, phi_w,   phi_b,   Phi);
  conv_kernel<2><<<dim3(TKW / 128, NB), 256, 0, stream>>>(xp, g_w,     g_b,     Gt);
  attn_kernel<<<dim3(THW / 128, NB), 256, 0, stream>>>(Qbuf, Phi, Gt, Ybuf);
  proj_kernel<<<dim3(THW / 64, OCH / 128, NB), 256, 0, stream>>>(Ybuf, W2h, W_b, out);
}
